// HypergraphEdgeAttentionBlock_28286654612014
// MI455X (gfx1250) — matrix operand order repaired and checked by reading, not compiled, not run on hardware
//
#include <hip/hip_runtime.h>
#include <hip/hip_bf16.h>

// ---------------------------------------------------------------------------
// Fused hypergraph edge-attention block for gfx1250 (CDNA5, wave32, WMMA).
// All GEMM-like work (km table, qm projection, both MLP layers) runs on
// v_wmma_f32_16x16x32_f16. M=64 edges/block so WMMA B-fragments are held in
// VGPRs and reused across 4 row tiles (4x less L2 weight traffic).
// ---------------------------------------------------------------------------

typedef __attribute__((ext_vector_type(16))) _Float16 v16h;
typedef __attribute__((ext_vector_type(8)))  _Float16 v8h;
typedef __attribute__((ext_vector_type(8)))  float    v8f;

// A-fragment (16x32 f16, M=lane&15): halves 0..15 = K k0+hi*16 .. +15
// (contiguous, matching B-fragment K order)
__device__ __forceinline__ v16h load_a16(const _Float16* p) {
    v16h a;
    *(v8h*)&a               = *(const v8h*)p;
    *(v8h*)((char*)&a + 16) = *(const v8h*)(p + 8);
    return a;
}
// B-fragment (32x16 f16, N=lane&15): low half-wave K k0..k0+15,
// high half-wave K k0+16..k0+31 -> 16 contiguous halves per lane.
__device__ __forceinline__ v16h load_b16(const _Float16* p) {
    v16h b;
    *(v8h*)&b               = *(const v8h*)p;
    *(v8h*)((char*)&b + 16) = *(const v8h*)(p + 8);
    return b;
}

// ---------------------------------------------------------------------------
// Prep 1: fold Wq@Wmh, Wk@Wmh -> f16 transposed B-layout [16 n][128 k],
// effective biases, prefix-sum of num.
// ---------------------------------------------------------------------------
__global__ void prep_fold_kernel(const float* __restrict__ Wq, const float* __restrict__ bq,
                                 const float* __restrict__ Wk, const float* __restrict__ bk,
                                 const float* __restrict__ Wmh, const float* __restrict__ bmh,
                                 const int* __restrict__ num, int Bn,
                                 _Float16* __restrict__ WqEfft, _Float16* __restrict__ WkEfft,
                                 float* __restrict__ bqEff, float* __restrict__ bkEff,
                                 int* __restrict__ cum) {
    int t = threadIdx.x;   // 128 threads, t = d (input feature)
    for (int h = 0; h < 16; ++h) {
        float sq = 0.f, sk = 0.f;
        if (h < 8) {
            for (int a = 0; a < 8; ++a) {
                float w = Wmh[a * 8 + h];
                sq += Wq[t * 8 + a] * w;
                sk += Wk[t * 8 + a] * w;
            }
        }
        WqEfft[h * 128 + t] = (_Float16)sq;
        WkEfft[h * 128 + t] = (_Float16)sk;
    }
    if (t < 8) {
        float sq = 0.f, sk = 0.f;
        for (int a = 0; a < 8; ++a) {
            float w = Wmh[a * 8 + t];
            sq += bq[a] * w;
            sk += bk[a] * w;
        }
        bqEff[t] = sq + bmh[t];
        bkEff[t] = sk + bmh[t];
    }
    if (t == 0) {
        int acc = 0;
        int bn = Bn > 256 ? 256 : Bn;
        for (int b = 0; b < bn; ++b) { cum[b] = acc; acc += num[b]; }
        cum[bn] = acc;
    }
}

// ---------------------------------------------------------------------------
// Prep 2/3: transpose MLP weights to f16, K-padded, for direct WMMA-B loads.
// ---------------------------------------------------------------------------
__global__ void w1t_kernel(const float* __restrict__ W1, _Float16* __restrict__ W1t) {
    int i = blockIdx.x * 256 + threadIdx.x;           // 256*224 total
    if (i >= 256 * 224) return;
    int n = i / 224, k = i % 224;
    W1t[i] = (k < 200) ? (_Float16)W1[(size_t)k * 256 + n] : (_Float16)0.f;
}
__global__ void w2t_kernel(const float* __restrict__ W2, _Float16* __restrict__ W2t) {
    int i = blockIdx.x * 256 + threadIdx.x;           // 128*256 total
    if (i >= 128 * 256) return;
    int n = i >> 8, k = i & 255;
    W2t[i] = (_Float16)W2[(size_t)k * 128 + n];
}

// ---------------------------------------------------------------------------
// Prep 4: km[N][8] = nodes @ WkEff + bkEff via WMMA (64 nodes / 128 threads)
// ---------------------------------------------------------------------------
__global__ void __launch_bounds__(128)
km_kernel(const float* __restrict__ nodes, const _Float16* __restrict__ WkEfft,
          const float* __restrict__ bkEff, float* __restrict__ km, int N) {
    __shared__ __align__(16) _Float16 sNd[64 * 136];
    __shared__ float sbk[8];
    const int t  = threadIdx.x;
    const int n0 = blockIdx.x * 64;
    if (t < 8) sbk[t] = bkEff[t];
    for (int i = t; i < 64 * 128; i += 128) {
        int r = i >> 7, c = i & 127;
        int n = n0 + r;
        sNd[r * 136 + c] = (n < N) ? (_Float16)nodes[(size_t)n * 128 + c] : (_Float16)0.f;
    }
    __syncthreads();
    const int lane = t & 31, wave = t >> 5;
    const int m = lane & 15, hi = lane >> 4, nIdx = lane & 15;
    const int rt = wave;                               // row tile 0..3
    const _Float16* arow = &sNd[(rt * 16 + m) * 136 + hi * 16];
    v8f acc = {};
    for (int s = 0; s < 4; ++s) {
        v16h a = load_a16(arow + 32 * s);
        v16h b = load_b16(WkEfft + (size_t)nIdx * 128 + hi * 16 + 32 * s);
        acc = __builtin_amdgcn_wmma_f32_16x16x32_f16(false, a, false, b, (short)0, acc,
                                                     false, false);
    }
    if (nIdx < 8) {
        float bias = sbk[nIdx];
        for (int r2 = 0; r2 < 8; ++r2) {
            int row = rt * 16 + r2 + hi * 8;           // D: VGPR r -> M=r / M=8+r
            int n = n0 + row;
            if (n < N) km[(size_t)n * 8 + nIdx] = acc[r2] + bias;
        }
    }
}

// ---------------------------------------------------------------------------
// Main fused kernel: 64 edges per block, 256 threads (8 waves).
// ---------------------------------------------------------------------------
__global__ void __launch_bounds__(256)
edge_block_kernel(const float* __restrict__ feat, const int* __restrict__ ind,
                  const float* __restrict__ globals_,
                  const int* __restrict__ cum, const float* __restrict__ km,
                  const _Float16* __restrict__ WqEfft, const float* __restrict__ bqEff,
                  const _Float16* __restrict__ W1t, const _Float16* __restrict__ W2t,
                  const float* __restrict__ b1, const float* __restrict__ b2,
                  const float* __restrict__ gamma, const float* __restrict__ beta,
                  float* __restrict__ out, int E, int Bn) {
    __shared__ __align__(16) _Float16 sX[64 * 232];    // 29696 B  x = [feat|agg|g], K-pad 224
    __shared__ __align__(16) _Float16 sH[64 * 264];    // 33792 B  hidden (A for GEMM2)
    __shared__ float sO[64 * 132];                     // 33792 B  o (pre-LN)
    __shared__ float sQm[64 * 12];                     //  3072 B  qm tile
    __shared__ int   sInd[64 * 8];                     //  2048 B
    __shared__ int   sCum[257];
    __shared__ float sb1[256], sb2[128], sg[128], sbe[128], sbq[8];
    __shared__ float sMu[64], sRs[64];

    const int t  = threadIdx.x;                        // 256
    const int e0 = blockIdx.x * 64;
    const int lane = t & 31, wave = t >> 5;
    const int m = lane & 15, hi = lane >> 4, nIdx = lane & 15;

    // ---- stage parameters + feat tile (f16 into x) ------------------------
    if (t < 256) sb1[t] = b1[t];
    if (t < 128) { sb2[t] = b2[t]; sg[t] = gamma[t]; sbe[t] = beta[t]; }
    if (t < 8)   sbq[t] = bqEff[t];
    int bn = Bn > 256 ? 256 : Bn;
    for (int i = t; i <= bn; i += 256) sCum[i] = cum[i];
    for (int i = t; i < 64 * 128; i += 256) {
        int r = i >> 7, c = i & 127;
        int e = e0 + r;
        float v = (e < E) ? feat[(size_t)e * 128 + c] : 0.f;
        sX[r * 232 + c] = (_Float16)v;
    }
    __syncthreads();

    // ---- phase A: waves 0-3 -> qm via WMMA; waves 4-7 -> staging ----------
    if (wave < 4) {
        const int rt = wave;                           // row tile 0..3
        const _Float16* arow = &sX[(rt * 16 + m) * 232 + hi * 16];
        v8f acc = {};
        for (int s = 0; s < 4; ++s) {                  // K = 128
            v16h a = load_a16(arow + 32 * s);
            v16h b = load_b16(WqEfft + (size_t)nIdx * 128 + hi * 16 + 32 * s);
            acc = __builtin_amdgcn_wmma_f32_16x16x32_f16(false, a, false, b, (short)0, acc,
                                                         false, false);
        }
        if (nIdx < 8) {
            float bias = sbq[nIdx];
            for (int r2 = 0; r2 < 8; ++r2) {
                int row = rt * 16 + r2 + hi * 8;
                sQm[row * 12 + nIdx] = acc[r2] + bias;
            }
        }
    } else {
        const int t2 = t - 128;                        // 0..127
        // member-node indices
        for (int i = t2; i < 64 * 8; i += 128) {
            int e = e0 + (i >> 3);
            sInd[i] = (e < E) ? ind[(size_t)e * 8 + (i & 7)] : 0;
        }
        // per-graph globals -> x[:,136..199]  (thread owns one row-half)
        {
            int r = t2 >> 1, c0 = (t2 & 1) * 32;
            int e = e0 + r;
            int seg = 0;
            if (e < E) {
                int lo = 0, hb = bn - 1;
                while (lo < hb) { int mid = (lo + hb + 1) >> 1; if (sCum[mid] <= e) lo = mid; else hb = mid - 1; }
                seg = lo;
            }
            const float* g = &globals_[(size_t)seg * 64];
            for (int j = 0; j < 32; ++j) {
                float v = (e < E) ? g[c0 + j] : 0.f;
                sX[r * 232 + 136 + c0 + j] = (_Float16)v;
            }
        }
        // zero K-pad 200..223
        for (int i = t2; i < 64 * 24; i += 128) {
            int r = i / 24, c = 200 + i % 24;
            sX[r * 232 + c] = (_Float16)0.f;
        }
    }
    __syncthreads();

    // ---- attention: softmax over K=8 members, per (edge,head) -------------
    for (int it = 0; it < 2; ++it) {
        int idx = t + it * 256;                        // 512 = 64 rows * 8 heads
        int r = idx >> 3, h = idx & 7;
        int e = e0 + r;
        float agg = 0.f;
        if (e < E) {
            float q = sQm[r * 12 + h];
            float kg[8], mx = -1e30f;
            for (int k2 = 0; k2 < 8; ++k2) {
                int nd = sInd[r * 8 + k2];
                kg[k2] = km[(size_t)nd * 8 + h];
                mx = fmaxf(mx, q * kg[k2]);
            }
            float s = 0.f, a = 0.f;
            for (int k2 = 0; k2 < 8; ++k2) {
                float w = __expf(q * kg[k2] - mx);
                s += w; a += w * kg[k2];
            }
            agg = a / s;
        }
        sX[r * 232 + 128 + h] = (_Float16)agg;
    }
    __syncthreads();

    // ---- GEMM1: h = relu(x @ W1 + b1)   M=64 K=224 N=256 ------------------
    // wave owns N-tiles {2w, 2w+1}; B-fragments held in VGPRs across 4 row tiles
    for (int j = 0; j < 2; ++j) {
        int ct = wave * 2 + j;                         // 0..15
        v16h bf[7];
        const _Float16* bcol = W1t + (size_t)(ct * 16 + nIdx) * 224 + hi * 16;
        for (int s = 0; s < 7; ++s) bf[s] = load_b16(bcol + 32 * s);
        float bias = sb1[ct * 16 + nIdx];
        for (int rt = 0; rt < 4; ++rt) {
            const _Float16* arow = &sX[(rt * 16 + m) * 232 + hi * 16];
            v8f acc = {};
            for (int s = 0; s < 7; ++s) {
                v16h a = load_a16(arow + 32 * s);
                acc = __builtin_amdgcn_wmma_f32_16x16x32_f16(false, a, false, bf[s],
                                                             (short)0, acc, false, false);
            }
            for (int r2 = 0; r2 < 8; ++r2) {
                int row = rt * 16 + r2 + hi * 8;
                float v = fmaxf(acc[r2] + bias, 0.f);
                sH[row * 264 + ct * 16 + nIdx] = (_Float16)v;
            }
        }
    }
    __syncthreads();

    // ---- GEMM2: o = sigmoid(h @ W2 + b2)   M=64 K=256 N=128 ---------------
    {
        int ct = wave;                                 // 0..7
        v16h bf[8];
        const _Float16* bcol = W2t + (size_t)(ct * 16 + nIdx) * 256 + hi * 16;
        for (int s = 0; s < 8; ++s) bf[s] = load_b16(bcol + 32 * s);
        float bias = sb2[ct * 16 + nIdx];
        for (int rt = 0; rt < 4; ++rt) {
            const _Float16* arow = &sH[(rt * 16 + m) * 264 + hi * 16];
            v8f acc = {};
            for (int s = 0; s < 8; ++s) {
                v16h a = load_a16(arow + 32 * s);
                acc = __builtin_amdgcn_wmma_f32_16x16x32_f16(false, a, false, bf[s],
                                                             (short)0, acc, false, false);
            }
            for (int r2 = 0; r2 < 8; ++r2) {
                int row = rt * 16 + r2 + hi * 8;
                float v = acc[r2] + bias;
                sO[row * 132 + ct * 16 + nIdx] = 1.f / (1.f + __expf(-v));
            }
        }
    }
    __syncthreads();

    // ---- LayerNorm over 128 (population var, eps = 1e-3) ------------------
    if (t < 64) {
        const float* row = &sO[t * 132];
        float s = 0.f;
        for (int c = 0; c < 128; ++c) s += row[c];
        float mu = s * (1.f / 128.f);
        float v = 0.f;
        for (int c = 0; c < 128; ++c) { float d = row[c] - mu; v += d * d; }
        sMu[t] = mu;
        sRs[t] = rsqrtf(v * (1.f / 128.f) + 1e-3f);
    }
    __syncthreads();
    {
        int c = t & 127, r0 = (t >> 7) * 32;           // coalesced stores
        float gv = sg[c], bv = sbe[c];
        for (int r2 = 0; r2 < 32; ++r2) {
            int r = r0 + r2;
            int e = e0 + r;
            if (e < E)
                out[(size_t)e * 128 + c] = gv * (sO[r * 132 + c] - sMu[r]) * sRs[r] + bv;
        }
    }
}

// ---------------------------------------------------------------------------
extern "C" void kernel_launch(void* const* d_in, const int* in_sizes, int n_in,
                              void* d_out, int out_size, void* d_ws, size_t ws_size,
                              hipStream_t stream) {
    const float* nodes    = (const float*)d_in[0];
    const float* feat     = (const float*)d_in[1];
    const float* globals_ = (const float*)d_in[2];
    const int*   ind      = (const int*)d_in[3];
    const int*   num      = (const int*)d_in[4];
    const float* Wq  = (const float*)d_in[5];
    const float* bq  = (const float*)d_in[6];
    const float* Wk  = (const float*)d_in[7];
    const float* bk  = (const float*)d_in[8];
    const float* Wmh = (const float*)d_in[9];
    const float* bmh = (const float*)d_in[10];
    const float* W1  = (const float*)d_in[11];
    const float* b1  = (const float*)d_in[12];
    const float* W2  = (const float*)d_in[13];
    const float* b2  = (const float*)d_in[14];
    const float* gam = (const float*)d_in[15];
    const float* bet = (const float*)d_in[16];
    float* out = (float*)d_out;

    const int N  = in_sizes[0] / 128;
    const int E  = in_sizes[1] / 128;
    const int Bn = in_sizes[4];

    // workspace carve-up (32-B aligned)
    char* ws = (char*)d_ws;
    _Float16* WqEfft = (_Float16*)(ws + 0);      // 16*128*2 = 4096 B
    _Float16* WkEfft = (_Float16*)(ws + 4096);   // 4096 B
    float*    bqEff  = (float*)(ws + 8192);      // 32 B
    float*    bkEff  = (float*)(ws + 8224);      // 32 B
    int*      cum    = (int*)  (ws + 8256);      // 1028 B
    _Float16* W1t    = (_Float16*)(ws + 9472);   // 256*224*2 = 114688 B
    _Float16* W2t    = (_Float16*)(ws + 124160); // 128*256*2 = 65536 B
    float*    km     = (float*)(ws + 189696);    // N*8*4 B

    prep_fold_kernel<<<1, 128, 0, stream>>>(Wq, bq, Wk, bk, Wmh, bmh, num, Bn,
                                            WqEfft, WkEfft, bqEff, bkEff, cum);
    w1t_kernel<<<(256 * 224) / 256, 256, 0, stream>>>(W1, W1t);
    w2t_kernel<<<(128 * 256) / 256, 256, 0, stream>>>(W2, W2t);
    km_kernel<<<(N + 63) / 64, 128, 0, stream>>>(nodes, WkEfft, bkEff, km, N);
    edge_block_kernel<<<(E + 63) / 64, 256, 0, stream>>>(
        feat, ind, globals_, cum, km, WqEfft, bqEff, W1t, W2t,
        b1, b2, gam, bet, out, E, Bn);
}
